// Model_47991964566123
// MI455X (gfx1250) — matrix operand order repaired and checked by reading, not compiled, not run on hardware
//
#include <hip/hip_runtime.h>
#include <math.h>

typedef __attribute__((ext_vector_type(8)))  float         v8f;
typedef __attribute__((ext_vector_type(16))) __bf16        v16bf;
typedef __attribute__((ext_vector_type(4)))  unsigned int  v4u;

union FragU {
    struct { v4u lo, hi; } q;
    v16bf v;
};

__device__ __forceinline__ unsigned short f2bf(float f) {
    unsigned int u = __float_as_uint(f);
    u += 0x7FFFu + ((u >> 16) & 1u);          // round-to-nearest-even
    return (unsigned short)(u >> 16);
}

__device__ __forceinline__ unsigned int pack2bf(float a, float b) {
    return (unsigned int)f2bf(a) | ((unsigned int)f2bf(b) << 16);
}

__device__ __forceinline__ float wave_sum(float v) {
#pragma unroll
    for (int off = 16; off > 0; off >>= 1)
        v += __shfl_xor(v, off, 32);
    return v;
}

// ---------------------------------------------------------------------------
// GEMM: C[M,N] = act(A[M,K] @ B[K,N] + bias[N]) with bf16 WMMA, f32 accum.
// Block = 256 threads (8 waves). Block tile 128x128, K-step 32.
// Wave tile 32x64: 2 A frags x 4 B frags -> 8 WMMAs per K-step.
// Requires: N % 128 == 0, K % 32 == 0 (holds: H=512, H2=256, F=1280).
// ---------------------------------------------------------------------------
#define TM 128
#define TN 128
#define TK 32

__global__ __launch_bounds__(256)
void gemm_bf16_wmma(const float* __restrict__ A, const float* __restrict__ B,
                    const float* __restrict__ bias, float* __restrict__ C,
                    int M, int N, int K, int do_relu)
{
    __shared__ __align__(16) unsigned short As[TM * TK];  // [m][k]   8 KB
    __shared__ __align__(16) unsigned short Bs[TN * TK];  // [n][k]   8 KB

    const int tid  = threadIdx.x;
    const int lane = tid & 31;
    const int wv   = tid >> 5;     // 0..7
    const int wm   = wv & 3;       // 32-row slab
    const int wn   = wv >> 2;      // 64-col slab
    const int m0   = blockIdx.y * TM;
    const int n0   = blockIdx.x * TN;

    v8f acc[2][4];
#pragma unroll
    for (int i = 0; i < 2; ++i)
#pragma unroll
        for (int j = 0; j < 4; ++j)
            acc[i][j] = v8f{0.f,0.f,0.f,0.f,0.f,0.f,0.f,0.f};

    // A staging: thread -> (row 0..127, kc {0,16}), 16 consecutive floats
    const int a_row = tid >> 1;
    const int a_kc  = (tid & 1) * 16;
    // B staging: thread -> (n 0..127, kc {0,16}), 16 k-strided floats
    const int b_n   = tid & 127;
    const int b_kc  = (tid >> 7) * 16;

    for (int kb = 0; kb < K; kb += TK) {
        // ---- stage A tile (f32 -> packed bf16, 2x ds_store_b128) ----
        {
            const int gr = m0 + a_row;
            v4u d0, d1;
            if (gr < M) {
                const float4* src =
                    reinterpret_cast<const float4*>(A + (size_t)gr * K + kb + a_kc);
                float4 f0 = src[0], f1 = src[1], f2 = src[2], f3 = src[3];
                if (kb + TK < K) __builtin_prefetch(src + (size_t)TK / 4 * 0 + (TK / 4), 0, 1);
                d0[0] = pack2bf(f0.x, f0.y); d0[1] = pack2bf(f0.z, f0.w);
                d0[2] = pack2bf(f1.x, f1.y); d0[3] = pack2bf(f1.z, f1.w);
                d1[0] = pack2bf(f2.x, f2.y); d1[1] = pack2bf(f2.z, f2.w);
                d1[2] = pack2bf(f3.x, f3.y); d1[3] = pack2bf(f3.z, f3.w);
            } else {
                d0 = v4u{0u,0u,0u,0u}; d1 = v4u{0u,0u,0u,0u};
            }
            *reinterpret_cast<v4u*>(&As[a_row * TK + a_kc])     = d0;
            *reinterpret_cast<v4u*>(&As[a_row * TK + a_kc + 8]) = d1;
        }
        // ---- stage B tile transposed into [n][k] (coalesced per j) ----
        {
            const float* src = B + (size_t)(kb + b_kc) * N + n0 + b_n;
            float v[16];
#pragma unroll
            for (int j = 0; j < 16; ++j) v[j] = src[(size_t)j * N];
            if (kb + TK < K) __builtin_prefetch(src + (size_t)TK * N, 0, 1);
            v4u d0, d1;
#pragma unroll
            for (int j = 0; j < 4; ++j) d0[j] = pack2bf(v[2*j],   v[2*j+1]);
#pragma unroll
            for (int j = 0; j < 4; ++j) d1[j] = pack2bf(v[8+2*j], v[8+2*j+1]);
            *reinterpret_cast<v4u*>(&Bs[b_n * TK + b_kc])     = d0;
            *reinterpret_cast<v4u*>(&Bs[b_n * TK + b_kc + 8]) = d1;
        }
        __syncthreads();

        // ---- fragments ----
        // A: lane half-group kbase {0,8}; K interleave {0-7,16-23}/{8-15,24-31}
        const int kb8 = (lane >> 4) * 8;
        FragU fa[2];
#pragma unroll
        for (int mi = 0; mi < 2; ++mi) {
            const int ar = wm * 32 + mi * 16 + (lane & 15);
            fa[mi].q.lo = *reinterpret_cast<const v4u*>(&As[ar * TK + kb8]);
            fa[mi].q.hi = *reinterpret_cast<const v4u*>(&As[ar * TK + 16 + kb8]);
        }
        // B: same K interleave as A {0-7,16-23}/{8-15,24-31}
        const int ks = kb8;
        FragU fb[4];
#pragma unroll
        for (int ni = 0; ni < 4; ++ni) {
            const int bc = wn * 64 + ni * 16 + (lane & 15);
            fb[ni].q.lo = *reinterpret_cast<const v4u*>(&Bs[bc * TK + ks]);
            fb[ni].q.hi = *reinterpret_cast<const v4u*>(&Bs[bc * TK + ks + 16]);
        }

#pragma unroll
        for (int mi = 0; mi < 2; ++mi)
#pragma unroll
            for (int ni = 0; ni < 4; ++ni)
                acc[mi][ni] = __builtin_amdgcn_wmma_f32_16x16x32_bf16(
                    false, fa[mi].v, false, fb[ni].v,
                    (short)0, acc[mi][ni], false, false);
        __syncthreads();
    }

    // ---- epilogue: C/D layout -> row = i + 8*(lane>=16), col = lane&15 ----
#pragma unroll
    for (int mi = 0; mi < 2; ++mi) {
        const int rowBase = m0 + wm * 32 + mi * 16 + ((lane >> 4) << 3);
#pragma unroll
        for (int ni = 0; ni < 4; ++ni) {
            const int col = n0 + wn * 64 + ni * 16 + (lane & 15);
            const float bcol = bias[col];
#pragma unroll
            for (int i = 0; i < 8; ++i) {
                const int row = rowBase + i;
                if (row < M) {
                    float v = acc[mi][ni][i] + bcol;
                    if (do_relu) v = fmaxf(v, 0.f);
                    C[(size_t)row * N + col] = v;
                }
            }
        }
    }
}

// ---------------------------------------------------------------------------
// Per-node inverse L2 norm: one wave per node (H multiple of 128).
// ---------------------------------------------------------------------------
__global__ __launch_bounds__(256)
void rownorm_inv(const float* __restrict__ h, float* __restrict__ invn,
                 int N, int H)
{
    const int lane = threadIdx.x & 31;
    int node = blockIdx.x * 8 + (threadIdx.x >> 5);
    if (node >= N) return;
    const float4* row = reinterpret_cast<const float4*>(h + (size_t)node * H);
    float s = 0.f;
    const int nq = H >> 2;
    for (int i = lane; i < nq; i += 32) {
        float4 a = row[i];
        s += a.x * a.x + a.y * a.y + a.z * a.z + a.w * a.w;
    }
    s = wave_sum(s);
    if (lane == 0) invn[node] = 1.0f / fmaxf(sqrtf(s), 1e-12f);
}

// ---------------------------------------------------------------------------
// Per-edge attention logits (one wave per edge, incl. self loops e>=E):
//   ex[e] = exp(beta * cos(h[src], h[dst])); denom[dst] += ex[e]
// Max-subtraction dropped: logits bounded by |beta|.
// ---------------------------------------------------------------------------
__global__ __launch_bounds__(256)
void edge_logits(const float* __restrict__ h, const float* __restrict__ invn,
                 const int* __restrict__ ei, int E, int N, int H,
                 const float* __restrict__ beta_ptr, int use_beta,
                 float* __restrict__ ex, float* __restrict__ denom)
{
    const int lane = threadIdx.x & 31;
    const int Et   = E + N;
    int e = blockIdx.x * 8 + (threadIdx.x >> 5);
    if (e >= Et) return;
    const int src = (e < E) ? ei[e]     : (e - E);
    const int dst = (e < E) ? ei[E + e] : (e - E);

    const float4* rs = reinterpret_cast<const float4*>(h + (size_t)src * H);
    const float4* rd = reinterpret_cast<const float4*>(h + (size_t)dst * H);
    float s = 0.f;
    const int nq = H >> 2;
    for (int i = lane; i < nq; i += 32) {
        float4 a = rs[i], b = rd[i];
        s += a.x * b.x + a.y * b.y + a.z * b.z + a.w * b.w;
    }
    s = wave_sum(s);
    if (lane == 0) {
        const float beta  = use_beta ? beta_ptr[0] : 1.0f;
        const float alpha = beta * s * invn[src] * invn[dst];
        const float exv   = __expf(alpha);
        ex[e] = exv;
        atomicAdd(&denom[dst], exv);
    }
}

// ---------------------------------------------------------------------------
// Weighted scatter-add: out[dst] += (ex[e]/denom[dst]) * h[src]
// ---------------------------------------------------------------------------
__global__ __launch_bounds__(256)
void edge_scatter(const float* __restrict__ h, const int* __restrict__ ei,
                  const float* __restrict__ ex, const float* __restrict__ denom,
                  float* __restrict__ out, int E, int N, int H)
{
    const int lane = threadIdx.x & 31;
    const int Et   = E + N;
    int e = blockIdx.x * 8 + (threadIdx.x >> 5);
    if (e >= Et) return;
    const int src = (e < E) ? ei[e]     : (e - E);
    const int dst = (e < E) ? ei[E + e] : (e - E);

    float w = 0.f;
    if (lane == 0) w = ex[e] / denom[dst];
    w = __shfl(w, 0, 32);

    const float4* rs = reinterpret_cast<const float4*>(h + (size_t)src * H);
    float* od = out + (size_t)dst * H;
    const int nq = H >> 2;
    for (int i = lane; i < nq; i += 32) {
        float4 a = rs[i];
        const int b = i << 2;
        atomicAdd(&od[b + 0], w * a.x);
        atomicAdd(&od[b + 1], w * a.y);
        atomicAdd(&od[b + 2], w * a.z);
        atomicAdd(&od[b + 3], w * a.w);
    }
}

// ---------------------------------------------------------------------------
// Global max/mean pool per graph (relu'd inputs >= 0 -> uint atomicMax works)
// ---------------------------------------------------------------------------
__global__ __launch_bounds__(256)
void pool_kernel(const float* __restrict__ h2, const int* __restrict__ batch,
                 float* __restrict__ gmx, float* __restrict__ gsum,
                 float* __restrict__ cnt, int N, int H2)
{
    const int n = blockIdx.x;
    const int f = threadIdx.x;
    if (n >= N || f >= H2) return;
    const int g   = batch[n];
    const float v = h2[(size_t)n * H2 + f];
    atomicMax(reinterpret_cast<unsigned int*>(&gmx[(size_t)g * H2 + f]),
              __float_as_uint(v));
    atomicAdd(&gsum[(size_t)g * H2 + f], v);
    if (f == 0) atomicAdd(&cnt[g], 1.0f);
}

// ---------------------------------------------------------------------------
// Head: out[g,c] = concat(gmx, gsum/cnt)[g] @ W3[:,c] + b3[c]
// ---------------------------------------------------------------------------
__global__ __launch_bounds__(32)
void head_kernel(const float* __restrict__ gmx, const float* __restrict__ gsum,
                 const float* __restrict__ cnt, const float* __restrict__ W3,
                 const float* __restrict__ b3, float* __restrict__ out,
                 int G, int C, int H2)
{
    const int g = blockIdx.x;
    const int c = threadIdx.x;
    if (c >= C) return;
    const float invc = 1.0f / fmaxf(cnt[g], 1.0f);
    float s = b3[c];
    for (int k = 0; k < H2; ++k) {
        s += gmx[(size_t)g * H2 + k] * W3[(size_t)k * C + c];
        s += gsum[(size_t)g * H2 + k] * invc * W3[(size_t)(H2 + k) * C + c];
    }
    out[(size_t)g * C + c] = s;
}

// ---------------------------------------------------------------------------

static inline size_t align256(size_t x) { return (x + 255) & ~(size_t)255; }

extern "C" void kernel_launch(void* const* d_in, const int* in_sizes, int n_in,
                              void* d_out, int out_size, void* d_ws, size_t ws_size,
                              hipStream_t stream)
{
    const float* x     = (const float*)d_in[0];
    const int*   ei    = (const int*)  d_in[1];
    const int*   batch = (const int*)  d_in[2];
    const float* W1    = (const float*)d_in[3];
    const float* b1    = (const float*)d_in[4];
    const float* beta2 = (const float*)d_in[5];
    const float* W2    = (const float*)d_in[6];
    const float* b2    = (const float*)d_in[7];
    const float* W3    = (const float*)d_in[8];
    const float* b3    = (const float*)d_in[9];
    float* out = (float*)d_out;

    const int N  = in_sizes[2];
    const int E  = in_sizes[1] / 2;
    const int F  = in_sizes[0] / N;
    const int H  = in_sizes[4];
    const int H2 = in_sizes[7];
    const int C  = in_sizes[9];
    const int G  = out_size / C;
    const int Et = E + N;

    // workspace carve-up
    char* ws = (char*)d_ws;
    size_t off = 0;
    float* h1    = (float*)(ws + off); off = align256(off + (size_t)N * H * 4);
    float* hA    = (float*)(ws + off); off = align256(off + (size_t)N * H * 4);
    float* h2    = (float*)(ws + off); off = align256(off + (size_t)N * H2 * 4);
    float* invn  = (float*)(ws + off); off = align256(off + (size_t)N * 4);
    float* denom = (float*)(ws + off); off = align256(off + (size_t)N * 4);
    float* exbuf = (float*)(ws + off); off = align256(off + (size_t)Et * 4);
    float* gmx   = (float*)(ws + off); off = align256(off + (size_t)G * H2 * 4);
    float* gsum  = (float*)(ws + off); off = align256(off + (size_t)G * H2 * 4);
    float* cnt   = (float*)(ws + off); off = align256(off + (size_t)G * 4);
    if (off > ws_size) return;  // insufficient scratch; nothing safe to do

    const dim3 blk256(256);
    const int edgeBlocks = (Et + 7) / 8;
    const int nodeBlocks = (N + 7) / 8;

    // 1) h1 = relu(x @ W1 + b1)            [N,F]x[F,H]
    {
        dim3 grid(H / TN, (N + TM - 1) / TM);
        gemm_bf16_wmma<<<grid, blk256, 0, stream>>>(x, W1, b1, h1, N, H, F, 1);
    }

    // 2) prop1: hA = AGNN(h1), beta = 1
    hipMemsetAsync(hA, 0, (size_t)N * H * 4, stream);
    hipMemsetAsync(denom, 0, (size_t)N * 4, stream);
    rownorm_inv<<<nodeBlocks, blk256, 0, stream>>>(h1, invn, N, H);
    edge_logits<<<edgeBlocks, blk256, 0, stream>>>(h1, invn, ei, E, N, H,
                                                   beta2, 0, exbuf, denom);
    edge_scatter<<<edgeBlocks, blk256, 0, stream>>>(h1, ei, exbuf, denom, hA, E, N, H);

    // 3) prop2: h1 = AGNN(hA), beta = beta2
    hipMemsetAsync(h1, 0, (size_t)N * H * 4, stream);
    hipMemsetAsync(denom, 0, (size_t)N * 4, stream);
    rownorm_inv<<<nodeBlocks, blk256, 0, stream>>>(hA, invn, N, H);
    edge_logits<<<edgeBlocks, blk256, 0, stream>>>(hA, invn, ei, E, N, H,
                                                   beta2, 1, exbuf, denom);
    edge_scatter<<<edgeBlocks, blk256, 0, stream>>>(hA, ei, exbuf, denom, h1, E, N, H);

    // 4) h2 = relu(h1 @ W2 + b2)           [N,H]x[H,H2]
    {
        dim3 grid(H2 / TN, (N + TM - 1) / TM);
        gemm_bf16_wmma<<<grid, blk256, 0, stream>>>(h1, W2, b2, h2, N, H2, H, 1);
    }

    // 5) pooling
    hipMemsetAsync(gmx, 0, (size_t)G * H2 * 4, stream);
    hipMemsetAsync(gsum, 0, (size_t)G * H2 * 4, stream);
    hipMemsetAsync(cnt, 0, (size_t)G * 4, stream);
    pool_kernel<<<N, dim3(H2), 0, stream>>>(h2, batch, gmx, gsum, cnt, N, H2);

    // 6) head
    head_kernel<<<G, dim3(32), 0, stream>>>(gmx, gsum, cnt, W3, b3, out, G, C, H2);
}
